// LSTMSynthesis_40939628266121
// MI455X (gfx1250) — compile-verified, matrix operand order checked by reading, not run on hardware
//
#include <hip/hip_runtime.h>
#include <hip/hip_bf16.h>

// ---------------------------------------------------------------------------
// Graves handwriting-synthesis LSTM, B=64 T=600 U=80 V=60 CS=400 K=10 M=20.
// Strategy (MI455X): batch is 4 WMMA m-tiles -> recurrence is batch-parallel,
// one persistent workgroup per 16-row tile; f16 WMMA w/ f32 accum; weights
// pre-shuffled to fragment order (L2-resident, ~4.7MB); activations live in
// packed f16 rows (act1/act2) moved with async global<->LDS copies (ASYNCcnt).
// ---------------------------------------------------------------------------
#define BB   64
#define TT   600
#define UU   80
#define VV   60

typedef _Float16 v16h __attribute__((ext_vector_type(16)));
typedef _Float16 v8h  __attribute__((ext_vector_type(8)));
typedef float    v8f  __attribute__((ext_vector_type(8)));

// K layouts: act1 row (480 = 15kt): [0..2]=x [3..62]=w [63]=0 [64..463]=h1 [464..479]=0
//            phase2 A  (896 = 28kt): [act1 row | h2(400) | 0(16)]
//            outproj A (896 = 28kt): [act1 row | act2 row(416)]
#define KT1   15
#define KT2   28
#define NT12  100       /* 1600 gate cols / 16 */
#define NTO   8         /* 128 padded output cols / 16 */
#define NKW   13        /* Ww k-tiles: kt 2..14 cover k=64..479 */
#define ASTR1 496       /* LDS halves per A row: 16B aligned, bank staggered */
#define ASTR2 912

// ---- workspace layout (bytes, 256-aligned) --------------------------------
#define WS_W1F   ((size_t)0)            /* 15*100*512 h = 1,536,000 B */
#define WS_W2F   ((size_t)1536000)      /* 28*100*512 h = 2,867,200 B */
#define WS_WOF   ((size_t)4403200)      /* 28*8*512  h  =   229,376 B */
#define WS_WWF   ((size_t)4632576)      /* 13*2*512  h  =    26,624 B */
#define WS_ACT1  ((size_t)4659200)      /* 64*600*480 f16 = 36,864,000 B */
#define WS_ACT2  ((size_t)41523200)     /* 64*600*416 f16 = 31,948,800 B */
#define WS_PAR   ((size_t)73472000)     /* 38400*128 f32  = 19,660,800 B */

// ---- d_out layout (f32 elements, reference return order) ------------------
#define OUT_END   ((size_t)0)
#define OUT_WGT   ((size_t)38400)
#define OUT_MU1   ((size_t)806400)
#define OUT_MU2   ((size_t)1574400)
#define OUT_LS1   ((size_t)2342400)
#define OUT_LS2   ((size_t)3110400)
#define OUT_RHO   ((size_t)3878400)
#define OUT_WOLD  ((size_t)4646400)
#define OUT_KAP   ((size_t)4650240)
#define OUT_H1    ((size_t)4650880)
#define OUT_C1    ((size_t)4676480)
#define OUT_H2    ((size_t)4702080)
#define OUT_C2    ((size_t)4727680)
#define OUT_PHI   ((size_t)4753280)

// ---- CDNA5 async global<->LDS copies (ASYNCcnt), guarded ------------------
#if defined(__HIP_DEVICE_COMPILE__) &&                                        \
    __has_builtin(__builtin_amdgcn_global_load_async_to_lds_b128) &&          \
    __has_builtin(__builtin_amdgcn_global_store_async_from_lds_b128) &&       \
    __has_builtin(__builtin_amdgcn_s_wait_asynccnt)
#define HAVE_ASYNC 1
#else
#define HAVE_ASYNC 0
#endif

#if HAVE_ASYNC
// Builtin prototype (from hipcc diagnostic): pointee is int __vector_size__(16),
// global operand in AS(1) (prints as __device__), LDS operand in AS(3).
typedef int v4i __attribute__((vector_size(16)));
typedef __attribute__((address_space(1))) v4i gv4i;
typedef __attribute__((address_space(3))) v4i lv4i;
__device__ __forceinline__ void cp_g2l16(void* lds, const void* g) {
  // 16B async copy global -> LDS, tracked by ASYNCcnt
  __builtin_amdgcn_global_load_async_to_lds_b128(
      (gv4i*)(unsigned long long)(size_t)g,
      (lv4i*)(unsigned int)(size_t)lds, 0, 0);
}
__device__ __forceinline__ void cp_l2g16(void* g, const void* lds) {
  // 16B async copy LDS -> global, tracked by ASYNCcnt
  __builtin_amdgcn_global_store_async_from_lds_b128(
      (gv4i*)(unsigned long long)(size_t)g,
      (lv4i*)(unsigned int)(size_t)lds, 0, 0);
}
__device__ __forceinline__ void async_wait0() { __builtin_amdgcn_s_wait_asynccnt(0); }
#else
__device__ __forceinline__ void cp_g2l16(void* lds, const void* g) { *(uint4*)lds = *(const uint4*)g; }
__device__ __forceinline__ void cp_l2g16(void* g, const void* lds) { *(uint4*)g = *(const uint4*)lds; }
__device__ __forceinline__ void async_wait0() {}
#endif

__device__ __forceinline__ float sigf(float v) { return 1.0f / (1.0f + __expf(-v)); }

__device__ __forceinline__ v8f wmma16(v16h a, v16h b, v8f c) {
  return __builtin_amdgcn_wmma_f32_16x16x32_f16(false, a, false, b, (short)0, c, false, false);
}
__device__ __forceinline__ v16h load_a(const _Float16* rowp) {   // 2x ds_load_b128
  union { v16h v; v8h h[2]; } u;
  u.h[0] = *(const v8h*)(rowp);
  u.h[1] = *(const v8h*)(rowp + 16);
  return u.v;
}
__device__ __forceinline__ v16h load_b(const _Float16* p) {      // 2x global_load_b128
  union { v16h v; v8h h[2]; } u;
  u.h[0] = *(const v8h*)(p);
  u.h[1] = *(const v8h*)(p + 8);
  return u.v;
}

// ---------------------------------------------------------------------------
// Weight pre-shuffle: frag f = kt*NT + nt; half index f*512 + lane*16 + e with
// (k,n) = (kt*32 + e + 8*(lane>>4) + 8*(e>>3), nt*16 + (lane&15)).
// ---------------------------------------------------------------------------
__global__ void prep_w1_kernel(const float* __restrict__ W1_ih, const float* __restrict__ W1_hh,
                               _Float16* __restrict__ W1f) {
  int tid = blockIdx.x * 256 + threadIdx.x;
  if (tid >= KT1 * NT12 * 512) return;
  int e = tid & 15, lane = (tid >> 4) & 31, f = tid >> 9;
  int nt = f % NT12, kt = f / NT12;
  int k = kt * 32 + e + 8 * (lane >> 4) + 8 * (e >> 3);
  int n = nt * 16 + (lane & 15);
  float v = 0.f;
  if (k < 63)                  v = W1_ih[n * 63 + k];
  else if (k >= 64 && k < 464) v = W1_hh[n * 400 + (k - 64)];
  W1f[tid] = (_Float16)v;
}

__global__ void prep_w2_kernel(const float* __restrict__ W2_ih, const float* __restrict__ W2_hh,
                               _Float16* __restrict__ W2f) {
  int tid = blockIdx.x * 256 + threadIdx.x;
  if (tid >= KT2 * NT12 * 512) return;
  int e = tid & 15, lane = (tid >> 4) & 31, f = tid >> 9;
  int nt = f % NT12, kt = f / NT12;
  int k = kt * 32 + e + 8 * (lane >> 4) + 8 * (e >> 3);
  int n = nt * 16 + (lane & 15);
  float v = 0.f;
  if (k < 63)                   v = W2_ih[n * 463 + k];
  else if (k >= 64 && k < 464)  v = W2_ih[n * 463 + (k - 1)];
  else if (k >= 480 && k < 880) v = W2_hh[n * 400 + (k - 480)];
  W2f[tid] = (_Float16)v;
}

__global__ void prep_wo_kernel(const float* __restrict__ Wo, _Float16* __restrict__ Wof) {
  int tid = blockIdx.x * 256 + threadIdx.x;
  if (tid >= KT2 * NTO * 512) return;
  int e = tid & 15, lane = (tid >> 4) & 31, f = tid >> 9;
  int nt = f % NTO, kt = f / NTO;
  int k = kt * 32 + e + 8 * (lane >> 4) + 8 * (e >> 3);
  int n = nt * 16 + (lane & 15);
  float v = 0.f;
  if (n < 121) {
    if (k >= 64 && k < 464)       v = Wo[n * 800 + (k - 64)];        // h1 part
    else if (k >= 480 && k < 880) v = Wo[n * 800 + 400 + (k - 480)]; // h2 part
  }
  Wof[tid] = (_Float16)v;
}

__global__ void prep_ww_kernel(const float* __restrict__ Ww, _Float16* __restrict__ WwF) {
  int tid = blockIdx.x * 256 + threadIdx.x;
  if (tid >= NKW * 2 * 512) return;
  int e = tid & 15, lane = (tid >> 4) & 31, f = tid >> 9;
  int nt = f & 1, kt = (f >> 1) + 2;                 // kt 2..14 -> k 64..479
  int k = kt * 32 + e + 8 * (lane >> 4) + 8 * (e >> 3);
  int n = nt * 16 + (lane & 15);
  float v = 0.f;
  if (n < 30 && k >= 64 && k < 464) v = Ww[n * 400 + (k - 64)];
  WwF[tid] = (_Float16)v;
}

// ---------------------------------------------------------------------------
// Phase 1: LSTM1 + attention, persistent over T. Grid = 4 x 256 (8 waves).
// Ping-pong A buffers: GEMM reads Acur=[x_t|w_{t-1}|h_{t-1}]; cell writes h_t
// and attention writes w_t into Anxt; Anxt row == act1[t] (async-stored).
// p = exp(h1@Ww.T+bw) is its own WMMA on the fresh h-part of Anxt.
// ---------------------------------------------------------------------------
__global__ __launch_bounds__(256) void phase1_kernel(
    const float* __restrict__ x, const float* __restrict__ onehots,
    const float* __restrict__ text_lens,
    const float* __restrict__ w0, const float* __restrict__ kap0,
    const float* __restrict__ h0, const float* __restrict__ c0,
    const float* __restrict__ b1, const float* __restrict__ bw,
    const _Float16* __restrict__ W1f, const _Float16* __restrict__ WwF,
    _Float16* __restrict__ act1, float* __restrict__ out)
{
  __shared__ _Float16 A0[16 * ASTR1];
  __shared__ _Float16 A1[16 * ASTR1];
  __shared__ float wL[16 * 64];
  __shared__ float kapL[16 * 16];
  __shared__ float pL[16 * 32];
  __shared__ float phiL[16 * 88];

  const int tid  = threadIdx.x;
  const int wave = tid >> 5, lane = tid & 31;
  const int hi   = lane >> 4, ln = lane & 15;
  const int b0   = blockIdx.x * 16;

  // ---- initial state: stage A0 = [x_0 | w_init | 0 | h_init | 0] ----
  for (int i = tid; i < 16 * 480; i += 256) {
    int r = i / 480, c = i % 480;
    float v;
    if (c < 3)        v = x[((size_t)(b0 + r) * TT + 0) * 3 + c];
    else if (c < 63)  v = w0[(b0 + r) * VV + (c - 3)];
    else if (c < 64)  v = 0.f;
    else if (c < 464) v = h0[(b0 + r) * 400 + (c - 64)];
    else              v = 0.f;
    A0[r * ASTR1 + c] = (_Float16)v;
  }
  for (int i = tid; i < 16; i += 256)      A1[i * ASTR1 + 63] = (_Float16)0.f;
  for (int i = tid; i < 16 * 16; i += 256) A1[(i >> 4) * ASTR1 + 464 + (i & 15)] = (_Float16)0.f;
  for (int i = tid; i < 16 * 64; i += 256) { int r = i >> 6, c = i & 63; wL[i] = (c < VV) ? w0[(b0 + r) * VV + c] : 0.f; }
  for (int i = tid; i < 16 * 16; i += 256) { int r = i >> 4, c = i & 15; kapL[i] = (c < 10) ? kap0[(b0 + r) * 10 + c] : 0.f; }
  float creg[4][8];
  #pragma unroll
  for (int q = 0; q < 4; ++q) {
    int ht = wave + 8 * q;
    if (ht < 25) {
      int j = ht * 16 + ln;
      #pragma unroll
      for (int r = 0; r < 8; ++r) creg[q][r] = c0[(b0 + r + 8 * hi) * 400 + j];
    }
  }
  __syncthreads();

  for (int t = 0; t < TT; ++t) {
    _Float16* Ac = (t & 1) ? A1 : A0;
    _Float16* An = (t & 1) ? A0 : A1;

    // ---- gate GEMM (K=480) fused with LSTM cell; h_t -> Anxt ----
    #pragma unroll
    for (int q = 0; q < 4; ++q) {
      int ht = wave + 8 * q;
      if (ht < 25) {
        v8f ai = {0.f,0.f,0.f,0.f,0.f,0.f,0.f,0.f};
        v8f af = ai, ag = ai, ao = ai;
        for (int kt = 0; kt < KT1; ++kt) {
          v16h av = load_a(&Ac[ln * ASTR1 + kt * 32 + 8 * hi]);
          const _Float16* wb = W1f + (size_t)(kt * NT12) * 512 + (size_t)lane * 16;
          if (kt + 1 < KT1) __builtin_prefetch(wb + (size_t)NT12 * 512, 0, 1);
          ai = wmma16(av, load_b(wb + (size_t)(ht     ) * 512), ai);
          af = wmma16(av, load_b(wb + (size_t)(25 + ht) * 512), af);
          ag = wmma16(av, load_b(wb + (size_t)(50 + ht) * 512), ag);
          ao = wmma16(av, load_b(wb + (size_t)(75 + ht) * 512), ao);
        }
        int j = ht * 16 + ln;
        float gbi = b1[j], gbf = b1[400 + j], gbg = b1[800 + j], gbo = b1[1200 + j];
        #pragma unroll
        for (int r = 0; r < 8; ++r) {
          int m = r + 8 * hi;
          float cn = sigf(af[r] + gbf) * creg[q][r] + sigf(ai[r] + gbi) * tanhf(ag[r] + gbg);
          float hn = sigf(ao[r] + gbo) * tanhf(cn);
          creg[q][r] = cn;
          An[m * ASTR1 + 64 + j] = (_Float16)hn;      // h_t into next A
        }
      }
    }
    __syncthreads();

    // ---- p = exp(h_t @ Ww.T + bw) via WMMA (waves 0-1); others carry x_t ----
    if (wave < 2) {
      v8f acc = {0.f,0.f,0.f,0.f,0.f,0.f,0.f,0.f};
      for (int kt = 2; kt < KT1; ++kt) {
        v16h av = load_a(&An[ln * ASTR1 + kt * 32 + 8 * hi]);
        acc = wmma16(av, load_b(WwF + (size_t)((kt - 2) * 2 + wave) * 512 + (size_t)lane * 16), acc);
      }
      int jj = wave * 16 + ln;
      if (jj < 30) {
        float bwj = bw[jj];
        #pragma unroll
        for (int r = 0; r < 8; ++r) pL[(r + 8 * hi) * 32 + jj] = __expf(acc[r] + bwj);
      }
    } else {
      for (int i = tid - 64; i < 48; i += 192) {       // x_t carried into Anxt
        int r = i / 3, c = i % 3;
        An[r * ASTR1 + c] = Ac[r * ASTR1 + c];
      }
    }
    __syncthreads();

    for (int i = tid; i < 16 * 10; i += 256) {         // kappa += pre_kappa
      int r = i / 10, k = i % 10;
      kapL[r * 16 + k] += pL[r * 32 + 20 + k];
    }
    __syncthreads();
    for (int i = tid; i < 16 * 81; i += 256) {         // phi over U+1 indices
      int r = i / 81, u = i % 81;
      float s = 0.f;
      #pragma unroll
      for (int k = 0; k < 10; ++k) {
        float d = kapL[r * 16 + k] - (float)u;
        s += pL[r * 32 + k] * __expf(-pL[r * 32 + 10 + k] * d * d);
      }
      phiL[r * 88 + u] = s * (80.0f / text_lens[b0 + r]);
    }
    __syncthreads();
    for (int i = tid; i < 16 * VV; i += 256) {         // w_t = phi[:U] @ onehots
      int r = i / VV, j = i % VV;
      float s = 0.f;
      for (int u = 0; u < UU; ++u) s += phiL[r * 88 + u] * onehots[((size_t)(b0 + r) * UU + u) * VV + j];
      wL[r * 64 + j] = s;
      An[r * ASTR1 + 3 + j] = (_Float16)s;             // w_t into next A
    }
    __syncthreads();

    // ---- Anxt row == act1[t] = [x_t|w_t|0|h_t|0]: async LDS->global ----
    for (int i = tid; i < 16 * 60; i += 256) {
      int r = i / 60, ch = i % 60;
      cp_l2g16(act1 + ((size_t)(b0 + r) * TT + t) * 480 + ch * 8, &An[r * ASTR1 + ch * 8]);
    }
    async_wait0();
    __syncthreads();
    if (t + 1 < TT) {                                   // stage x_{t+1}
      for (int i = tid; i < 48; i += 256) {
        int r = i / 3, c = i % 3;
        An[r * ASTR1 + c] = (_Float16)x[((size_t)(b0 + r) * TT + (t + 1)) * 3 + c];
      }
    }
    __syncthreads();
  }

  // ---- final states (t = T-1 values live in A0 / LDS / regs) ----
  const _Float16* Af = A0;                              // TT even -> buf[TT&1]
  for (int i = tid; i < 16 * VV; i += 256)  { int r = i / VV,  j = i % VV;  out[OUT_WOLD + (b0 + r) * VV + j] = wL[r * 64 + j]; }
  for (int i = tid; i < 16 * 10; i += 256)  { int r = i / 10,  k = i % 10;  out[OUT_KAP + (b0 + r) * 10 + k] = kapL[r * 16 + k]; }
  for (int i = tid; i < 16 * 400; i += 256) { int r = i / 400, c = i % 400; out[OUT_H1 + (b0 + r) * 400 + c] = (float)Af[r * ASTR1 + 64 + c]; }
  for (int i = tid; i < 16 * 81; i += 256)  { int r = i / 81,  u = i % 81;  out[OUT_PHI + (b0 + r) * 81 + u] = phiL[r * 88 + u]; }
  #pragma unroll
  for (int q = 0; q < 4; ++q) {
    int ht = wave + 8 * q;
    if (ht < 25) {
      int j = ht * 16 + ln;
      #pragma unroll
      for (int r = 0; r < 8; ++r) out[OUT_C1 + (b0 + r + 8 * hi) * 400 + j] = creg[q][r];
    }
  }
}

// ---------------------------------------------------------------------------
// Phase 2: LSTM2. A = [act1[t] | h2 | 0] (K=896). act1[t+1] is async-prefetched
// into the next A buffer while the current GEMM runs; cell writes h2_t there.
// ---------------------------------------------------------------------------
__global__ __launch_bounds__(256) void phase2_kernel(
    const float* __restrict__ h2_0, const float* __restrict__ c2_0,
    const float* __restrict__ b2, const _Float16* __restrict__ W2f,
    const _Float16* __restrict__ act1, _Float16* __restrict__ act2,
    float* __restrict__ out)
{
  __shared__ _Float16 A0[16 * ASTR2];
  __shared__ _Float16 A1[16 * ASTR2];

  const int tid  = threadIdx.x;
  const int wave = tid >> 5, lane = tid & 31;
  const int hi   = lane >> 4, ln = lane & 15;
  const int b0   = blockIdx.x * 16;

  // ---- prologue: A0 <- [act1[0] | h2_init | 0]; zero pad cols of both bufs
  for (int i = tid; i < 16 * 60; i += 256) {
    int r = i / 60, ch = i % 60;
    cp_g2l16(&A0[r * ASTR2 + ch * 8], act1 + ((size_t)(b0 + r) * TT + 0) * 480 + ch * 8);
  }
  for (int i = tid; i < 16 * 400; i += 256) {
    int r = i / 400, c = i % 400;
    A0[r * ASTR2 + 480 + c] = (_Float16)h2_0[(b0 + r) * 400 + c];
  }
  for (int i = tid; i < 16 * 16; i += 256) {
    int r = i >> 4, c = i & 15;
    A0[r * ASTR2 + 880 + c] = (_Float16)0.f;
    A1[r * ASTR2 + 880 + c] = (_Float16)0.f;
  }
  float creg[4][8];
  #pragma unroll
  for (int q = 0; q < 4; ++q) {
    int ht = wave + 8 * q;
    if (ht < 25) {
      int j = ht * 16 + ln;
      #pragma unroll
      for (int r = 0; r < 8; ++r) creg[q][r] = c2_0[(b0 + r + 8 * hi) * 400 + j];
    }
  }
  async_wait0();
  __syncthreads();

  for (int t = 0; t < TT; ++t) {
    _Float16* Ac = (t & 1) ? A1 : A0;
    _Float16* An = (t & 1) ? A0 : A1;

    // async prefetch act1[t+1] into Anxt[0..479], overlapped with the GEMM
    if (t + 1 < TT) {
      for (int i = tid; i < 16 * 60; i += 256) {
        int r = i / 60, ch = i % 60;
        cp_g2l16(&An[r * ASTR2 + ch * 8], act1 + ((size_t)(b0 + r) * TT + (t + 1)) * 480 + ch * 8);
      }
    }

    #pragma unroll
    for (int q = 0; q < 4; ++q) {
      int ht = wave + 8 * q;
      if (ht < 25) {
        v8f ai = {0.f,0.f,0.f,0.f,0.f,0.f,0.f,0.f};
        v8f af = ai, ag = ai, ao = ai;
        for (int kt = 0; kt < KT2; ++kt) {
          v16h av = load_a(&Ac[ln * ASTR2 + kt * 32 + 8 * hi]);
          const _Float16* wb = W2f + (size_t)(kt * NT12) * 512 + (size_t)lane * 16;
          if (kt + 1 < KT2) __builtin_prefetch(wb + (size_t)NT12 * 512, 0, 1);
          ai = wmma16(av, load_b(wb + (size_t)(ht     ) * 512), ai);
          af = wmma16(av, load_b(wb + (size_t)(25 + ht) * 512), af);
          ag = wmma16(av, load_b(wb + (size_t)(50 + ht) * 512), ag);
          ao = wmma16(av, load_b(wb + (size_t)(75 + ht) * 512), ao);
        }
        int j = ht * 16 + ln;
        float gbi = b2[j], gbf = b2[400 + j], gbg = b2[800 + j], gbo = b2[1200 + j];
        #pragma unroll
        for (int r = 0; r < 8; ++r) {
          int m = r + 8 * hi;
          float cn = sigf(af[r] + gbf) * creg[q][r] + sigf(ai[r] + gbi) * tanhf(ag[r] + gbg);
          float hn = sigf(ao[r] + gbo) * tanhf(cn);
          creg[q][r] = cn;
          An[m * ASTR2 + 480 + j] = (_Float16)hn;                       // h2_t for next step
          act2[((size_t)(b0 + m) * TT + t) * 416 + j] = (_Float16)hn;   // h2_t for outproj
        }
      }
    }
    for (int i = tid; i < 16 * 16; i += 256) {          // act2 pad cols = 0
      int r = i >> 4, c = i & 15;
      act2[((size_t)(b0 + r) * TT + t) * 416 + 400 + c] = (_Float16)0.f;
    }
    async_wait0();
    __syncthreads();
  }

  const _Float16* Af = A0;                              // TT even
  for (int i = tid; i < 16 * 400; i += 256) { int r = i / 400, c = i % 400; out[OUT_H2 + (b0 + r) * 400 + c] = (float)Af[r * ASTR2 + 480 + c]; }
  #pragma unroll
  for (int q = 0; q < 4; ++q) {
    int ht = wave + 8 * q;
    if (ht < 25) {
      int j = ht * 16 + ln;
      #pragma unroll
      for (int r = 0; r < 8; ++r) out[OUT_C2 + (b0 + r + 8 * hi) * 400 + j] = creg[q][r];
    }
  }
}

// ---------------------------------------------------------------------------
// Output projection: params = [act1|act2] (38400 x 896) @ Wof (28 kt x 8 nt).
// A staged entirely by async global->LDS copies.
// ---------------------------------------------------------------------------
__global__ __launch_bounds__(256) void outproj_kernel(
    const _Float16* __restrict__ act1, const _Float16* __restrict__ act2,
    const _Float16* __restrict__ Wof, float* __restrict__ params)
{
  __shared__ _Float16 A[16 * ASTR2];
  const int tid  = threadIdx.x;
  const int wave = tid >> 5, lane = tid & 31;
  const int hi   = lane >> 4, ln = lane & 15;
  const size_t row0 = (size_t)blockIdx.x * 16;

  for (int i = tid; i < 16 * 112; i += 256) {          // 60 + 52 chunks / row
    int r = i / 112, ch = i % 112;
    size_t row = row0 + r;
    if (ch < 60) cp_g2l16(&A[r * ASTR2 + ch * 8], act1 + row * 480 + ch * 8);
    else { int c2 = ch - 60; cp_g2l16(&A[r * ASTR2 + 480 + c2 * 8], act2 + row * 416 + c2 * 8); }
  }
  async_wait0();
  __syncthreads();

  v8f acc = {0.f,0.f,0.f,0.f,0.f,0.f,0.f,0.f};
  const int nt = wave;
  for (int kt = 0; kt < KT2; ++kt) {
    v16h av = load_a(&A[ln * ASTR2 + kt * 32 + 8 * hi]);
    const _Float16* wb = Wof + (size_t)(kt * NTO + nt) * 512 + (size_t)lane * 16;
    if (kt + 1 < KT2) __builtin_prefetch(wb + (size_t)NTO * 512, 0, 1);
    acc = wmma16(av, load_b(wb), acc);
  }
  int n = nt * 16 + ln;
  #pragma unroll
  for (int r = 0; r < 8; ++r) params[(row0 + r + 8 * hi) * 128 + n] = acc[r];
}

// ---------------------------------------------------------------------------
__global__ __launch_bounds__(256) void finalize_kernel(
    const float* __restrict__ params, const float* __restrict__ bo,
    float* __restrict__ out)
{
  int row = blockIdx.x * 256 + threadIdx.x;
  if (row >= BB * TT) return;
  const float* pr = params + (size_t)row * 128;

  float pw[20], mx = -1e30f;
  #pragma unroll
  for (int i = 0; i < 20; ++i) { float v = pr[i] + bo[i]; pw[i] = v; mx = fmaxf(mx, v); }
  float se = 0.f;
  #pragma unroll
  for (int i = 0; i < 20; ++i) { pw[i] = __expf(pw[i] - mx); se += pw[i]; }
  float inv = 1.0f / se;

  out[OUT_END + row] = sigf(pr[120] + bo[120]);
  #pragma unroll
  for (int i = 0; i < 20; ++i) {
    out[OUT_WGT + (size_t)row * 20 + i] = pw[i] * inv;
    out[OUT_MU1 + (size_t)row * 20 + i] = pr[20 + i] + bo[20 + i];
    out[OUT_MU2 + (size_t)row * 20 + i] = pr[40 + i] + bo[40 + i];
    out[OUT_LS1 + (size_t)row * 20 + i] = pr[60 + i] + bo[60 + i];
    out[OUT_LS2 + (size_t)row * 20 + i] = pr[80 + i] + bo[80 + i];
    out[OUT_RHO + (size_t)row * 20 + i] = tanhf(pr[100 + i] + bo[100 + i]);
  }
}

// ---------------------------------------------------------------------------
extern "C" void kernel_launch(void* const* d_in, const int* in_sizes, int n_in,
                              void* d_out, int out_size, void* d_ws, size_t ws_size,
                              hipStream_t stream) {
  const float* x         = (const float*)d_in[0];
  const float* onehots   = (const float*)d_in[1];
  const float* text_lens = (const float*)d_in[2];
  const float* w_old     = (const float*)d_in[3];
  const float* kappa_old = (const float*)d_in[4];
  const float* h1_in     = (const float*)d_in[5];
  const float* c1_in     = (const float*)d_in[6];
  const float* h2_in     = (const float*)d_in[7];
  const float* c2_in     = (const float*)d_in[8];
  const float* W1_ih     = (const float*)d_in[9];
  const float* W1_hh     = (const float*)d_in[10];
  const float* b1        = (const float*)d_in[11];
  const float* Ww        = (const float*)d_in[12];
  const float* bw        = (const float*)d_in[13];
  const float* W2_ih     = (const float*)d_in[14];
  const float* W2_hh     = (const float*)d_in[15];
  const float* b2        = (const float*)d_in[16];
  const float* Wo        = (const float*)d_in[17];
  const float* bo        = (const float*)d_in[18];

  char* wsb = (char*)d_ws;
  _Float16* W1f  = (_Float16*)(wsb + WS_W1F);
  _Float16* W2f  = (_Float16*)(wsb + WS_W2F);
  _Float16* Wof  = (_Float16*)(wsb + WS_WOF);
  _Float16* WwF  = (_Float16*)(wsb + WS_WWF);
  _Float16* act1 = (_Float16*)(wsb + WS_ACT1);
  _Float16* act2 = (_Float16*)(wsb + WS_ACT2);
  float* params  = (float*)(wsb + WS_PAR);
  float* out     = (float*)d_out;

  prep_w1_kernel<<<(KT1 * NT12 * 512 + 255) / 256, 256, 0, stream>>>(W1_ih, W1_hh, W1f);
  prep_w2_kernel<<<(KT2 * NT12 * 512 + 255) / 256, 256, 0, stream>>>(W2_ih, W2_hh, W2f);
  prep_wo_kernel<<<(KT2 * NTO * 512 + 255) / 256, 256, 0, stream>>>(Wo, Wof);
  prep_ww_kernel<<<(NKW * 2 * 512 + 255) / 256, 256, 0, stream>>>(Ww, WwF);

  phase1_kernel<<<4, 256, 0, stream>>>(x, onehots, text_lens, w_old, kappa_old,
                                       h1_in, c1_in, b1, bw, W1f, WwF, act1, out);
  phase2_kernel<<<4, 256, 0, stream>>>(h2_in, c2_in, b2, W2f, act1, act2, out);

  outproj_kernel<<<(BB * TT) / 16, 256, 0, stream>>>(act1, act2, Wof, params);
  finalize_kernel<<<(BB * TT + 255) / 256, 256, 0, stream>>>(params, bo, out);
}
